// IPAttnProcessor_36043365548074
// MI455X (gfx1250) — compile-verified, matrix operand order checked by reading, not run on hardware
//
#include <hip/hip_runtime.h>
#include <math.h>

// ---------------------------------------------------------------------------
// Problem constants: B=4, S=4096, C=1280, Dc=2048, H=20, D=64, L=77+4+4=85.
// ---------------------------------------------------------------------------
#define BSZ   4
#define SEQ   4096
#define CDIM  1280
#define DCDIM 2048
#define HEADS 20
#define DHEAD 64
#define LTOT  85
#define LTXT  77
#define NIMG  4
#define NLBL  4
#define SCALE_IMG 1.0f
#define SCALE_LBL 1.0f

typedef __attribute__((ext_vector_type(16))) __bf16 v16bf;
typedef __attribute__((ext_vector_type(8)))  float  v8f;

// ---------------------------------------------------------------------------
// Async global->LDS copies (CDNA5 ASYNCcnt path), with sync fallback.
// Builtin signature (from clang diagnostic): arg0 = v2i addrspace(1)* (global
// source, printed as "__device__"), arg1 = v2i addrspace(3)* (LDS dest),
// then imm offset, imm cpol.
// ---------------------------------------------------------------------------
#if __has_builtin(__builtin_amdgcn_global_load_async_to_lds_b64)
#define HAVE_ASYNC_LDS 1
#else
#define HAVE_ASYNC_LDS 0
#endif

typedef int v2i __attribute__((vector_size(2 * sizeof(int))));
typedef __attribute__((address_space(1))) v2i* gv2i_p;   // 64-bit pointer
typedef __attribute__((address_space(3))) v2i* lv2i_p;   // 32-bit pointer

// copy 8 bytes global -> LDS
__device__ __forceinline__ void copy8_g2l(const unsigned short* g,
                                          unsigned short* l) {
#if HAVE_ASYNC_LDS
  gv2i_p gp = __builtin_bit_cast(gv2i_p, g);
  // generic shared addr = {aperture[63:32], lds_offset[31:0]} (ISA 10.2)
  lv2i_p lp = __builtin_bit_cast(lv2i_p, (unsigned int)(unsigned long long)l);
  __builtin_amdgcn_global_load_async_to_lds_b64(gp, lp, 0, 0);
#else
  *(uint2*)l = *(const uint2*)g;
#endif
}

__device__ __forceinline__ void wait_async_lds() {
#if HAVE_ASYNC_LDS
#if __has_builtin(__builtin_amdgcn_s_wait_asynccnt)
  __builtin_amdgcn_s_wait_asynccnt(0);
#else
  asm volatile("s_wait_asynccnt 0x0" ::: "memory");
#endif
#endif
}

// f32 -> bf16 round-to-nearest-even
__device__ __forceinline__ unsigned short f2bf(float f) {
  union { float f; unsigned int u; } x; x.f = f;
  unsigned int u = x.u;
  return (unsigned short)((u + 0x7FFFu + ((u >> 16) & 1u)) >> 16);
}

// Per-lane K index inside a 16x32 (16-bit) WMMA operand fragment.
// Lanes 0-15 hold K {0..7,16..23}; lanes 16-31 hold K {8..15,24..31}.
__device__ __forceinline__ int kmap16(int j, int hi) {
  int g = j >> 1;
  int base = (g < 4) ? (g << 1) : (16 + ((g - 4) << 1));
  return base + (j & 1) + (hi << 3);
}

__device__ __forceinline__ v16bf load_frag(const unsigned short* p, int hi) {
  v16bf f;
#pragma unroll
  for (int j = 0; j < 16; ++j)
    f[j] = __builtin_bit_cast(__bf16, p[kmap16(j, hi)]);
  return f;
}

__device__ __forceinline__ v8f v8f_zero() {
  v8f z;
#pragma unroll
  for (int e = 0; e < 8; ++e) z[e] = 0.0f;
  return z;
}

// ---------------------------------------------------------------------------
// Pre-pass 1: elementwise f32 -> bf16 (vectorized x4)
// ---------------------------------------------------------------------------
__global__ __launch_bounds__(256)
void cvt_bf16(const float* __restrict__ src, unsigned short* __restrict__ dst,
              long n4) {
  const long i = (long)blockIdx.x * 256 + threadIdx.x;
  if (i < n4) {
    const float4 v = ((const float4*)src)[i];
    uint2 o;
    o.x = (unsigned)f2bf(v.x) | ((unsigned)f2bf(v.y) << 16);
    o.y = (unsigned)f2bf(v.z) | ((unsigned)f2bf(v.w) << 16);
    ((uint2*)dst)[i] = o;
  }
}

// ---------------------------------------------------------------------------
// Pre-pass 2: W (KxN f32) -> Wt (NxK bf16), 32x32 LDS tiles.
// ---------------------------------------------------------------------------
__global__ __launch_bounds__(256)
void transpose_cvt(const float* __restrict__ W, unsigned short* __restrict__ Wt,
                   int K, int N) {
  __shared__ unsigned short t[32][33];
  const int tx = threadIdx.x & 31, ty = threadIdx.x >> 5;
  const int n0 = blockIdx.x * 32, k0 = blockIdx.y * 32;
#pragma unroll
  for (int rr = 0; rr < 4; ++rr) {
    const int k = ty + rr * 8;
    t[k][tx] = f2bf(W[(long)(k0 + k) * N + n0 + tx]);
  }
  __syncthreads();
#pragma unroll
  for (int rr = 0; rr < 4; ++rr) {
    const int n = ty + rr * 8;
    Wt[(long)(n0 + n) * K + k0 + tx] = t[tx][n];
  }
}

// ---------------------------------------------------------------------------
// bf16 GEMM on WMMA:  C = A(MxK,bf16) * Bt(NxK,bf16)^T  [+bias +resid -> f32]
// Row remap on A reads & C writes: grow = (m/Lseg)*Ltot + segOff + m%Lseg
// Block 128x128, 8 waves x (32x64). Double-buffered async global->LDS staging.
// LDS row stride 36 ush (72B): b64-aligned, 18-bank step => conflict-free
// fragment gathers. Requires N%128==0, K%32==0.
// ---------------------------------------------------------------------------
#define LDT 36

__global__ __launch_bounds__(256)
void gemm_bf16t(const unsigned short* __restrict__ A,
                const unsigned short* __restrict__ Bt,
                float* __restrict__ Cf,            // f32 out (or null)
                unsigned short* __restrict__ Cbf,  // bf16 out (or null)
                int Mseg, int N, int K,
                int Lseg, int Ltot, int segOff,
                const float* __restrict__ bias,
                const float* __restrict__ resid) {
  __shared__ unsigned short lA[2][128 * LDT];
  __shared__ unsigned short lB[2][128 * LDT];

  const int tid  = threadIdx.x;
  const int lane = tid & 31;
  const int wave = tid >> 5;
  const int r    = lane & 15;
  const int hi   = lane >> 4;
  const int mBase = blockIdx.x * 128;
  const int nBase = blockIdx.y * 128;
  const int waveM = (wave & 3) * 32;
  const int waveN = (wave >> 2) * 64;

  v8f acc[2][4];
#pragma unroll
  for (int i = 0; i < 2; ++i)
#pragma unroll
    for (int j = 0; j < 4; ++j) acc[i][j] = v8f_zero();

  // Staging: 128 rows x 32 k of bf16 = 1024 8B-chunks per tile; 4 per thread.
  const unsigned short* aG[4];
  const unsigned short* bG[4];
  int lOff[4];
#pragma unroll
  for (int i = 0; i < 4; ++i) {
    const int c = tid + i * 256;      // chunk id
    const int row = c >> 3, part = c & 7;
    int am = mBase + row;
    if (am >= Mseg) am = 0;           // clamp OOB rows (stores are guarded)
    const long aRow = (long)((am / Lseg) * Ltot + segOff + (am % Lseg));
    aG[i]   = A + aRow * K + part * 4;
    bG[i]   = Bt + (long)(nBase + row) * K + part * 4;
    lOff[i] = row * LDT + part * 4;
  }

  auto issue_tile = [&](int buf, int k0) {
#pragma unroll
    for (int i = 0; i < 4; ++i) copy8_g2l(aG[i] + k0, &lA[buf][lOff[i]]);
#pragma unroll
    for (int i = 0; i < 4; ++i) copy8_g2l(bG[i] + k0, &lB[buf][lOff[i]]);
  };

  issue_tile(0, 0);
  wait_async_lds();
  __syncthreads();

  const int nk = K >> 5;
  for (int kt = 0; kt < nk; ++kt) {
    const int cur = kt & 1, nxt = cur ^ 1;
    if (kt + 1 < nk) issue_tile(nxt, (kt + 1) * 32);

    v16bf af[2], bf[4];
#pragma unroll
    for (int fm = 0; fm < 2; ++fm)
      af[fm] = load_frag(&lA[cur][(waveM + fm * 16 + r) * LDT], hi);
#pragma unroll
    for (int fn = 0; fn < 4; ++fn)
      bf[fn] = load_frag(&lB[cur][(waveN + fn * 16 + r) * LDT], hi);

#pragma unroll
    for (int fm = 0; fm < 2; ++fm)
#pragma unroll
      for (int fn = 0; fn < 4; ++fn)
        acc[fm][fn] = __builtin_amdgcn_wmma_f32_16x16x32_bf16(
            false, af[fm], false, bf[fn], (short)0, acc[fm][fn],
            false, false);

    wait_async_lds();   // next tile resident before buffer swap
    __syncthreads();
  }

#pragma unroll
  for (int fm = 0; fm < 2; ++fm) {
#pragma unroll
    for (int fn = 0; fn < 4; ++fn) {
      const int col = nBase + waveN + fn * 16 + r;
#pragma unroll
      for (int vi = 0; vi < 8; ++vi) {
        const int m = mBase + waveM + fm * 16 + vi + hi * 8;
        if (m < Mseg) {
          const long grow = (long)((m / Lseg) * Ltot + segOff + (m % Lseg));
          float v = acc[fm][fn][vi];
          if (bias)  v += bias[col];
          if (resid) v += resid[grow * N + col];
          if (Cf)  Cf[grow * N + col]  = v;
          if (Cbf) Cbf[grow * N + col] = f2bf(v);
        }
      }
    }
  }
}

// ---------------------------------------------------------------------------
// Attention (all-bf16 inputs): out = sum_seg scale*softmax(q k^T/8) v
// One WG per (128 queries, head, batch); scores and P*V both on WMMA.
// Three per-segment softmaxes folded into one combined weight matrix P.
// ---------------------------------------------------------------------------
#define QSTR 68
#define KSTR 68
#define PSTR 100
#define VSTR 100

__global__ __launch_bounds__(256)
void attn_wmma(const unsigned short* __restrict__ Q,   // (B,S,C) bf16
               const unsigned short* __restrict__ Kb,  // (B,85,C) bf16
               const unsigned short* __restrict__ Vb,  // (B,85,C) bf16
               unsigned short* __restrict__ O) {       // (B,S,C) bf16
  __shared__ unsigned short lK[96 * KSTR];    // [key][d]
  __shared__ unsigned short lVT[64 * VSTR];   // [d][key]
  __shared__ unsigned short lQP[128 * PSTR];  // Q phase str 68; P phase str 100

  const int tid   = threadIdx.x;
  const int lane  = tid & 31;
  const int wave  = tid >> 5;
  const int r     = lane & 15;
  const int hi    = lane >> 4;
  const int h     = blockIdx.y;
  const int b     = blockIdx.z;
  const int qBase = blockIdx.x * 128;
  const int waveQ = wave * 16;
  const long cOff = (long)h * DHEAD;

  // stage Q (128x64), K (96x64 zero-padded), V^T (64x96 zero-padded)
#pragma unroll
  for (int i = 0; i < 8; ++i) {                 // 2048 4-ush chunks
    const int c = tid + i * 256;
    const int row = c >> 4, part = c & 15;
    const unsigned short* g =
        Q + ((long)b * SEQ + qBase + row) * CDIM + cOff + part * 4;
    *(uint2*)&lQP[row * QSTR + part * 4] = *(const uint2*)g;
  }
#pragma unroll
  for (int i = 0; i < 6; ++i) {                 // 1536 chunks
    const int c = tid + i * 256;
    const int key = c >> 4, part = c & 15;
    uint2 v; v.x = 0u; v.y = 0u;
    if (key < LTOT)
      v = *(const uint2*)(Kb + ((long)b * LTOT + key) * CDIM + cOff + part * 4);
    *(uint2*)&lK[key * KSTR + part * 4] = v;
  }
  for (int i = tid; i < 96 * 64; i += 256) {
    const int key = i >> 6, d = i & 63;
    unsigned short v = 0;
    if (key < LTOT) v = Vb[((long)b * LTOT + key) * CDIM + cOff + d];
    lVT[d * VSTR + key] = v;
  }
  __syncthreads();

  // scores: 16 queries x 96 keys, K-dim 64 (2 steps)
  v16bf aq[2];
#pragma unroll
  for (int ks = 0; ks < 2; ++ks)
    aq[ks] = load_frag(&lQP[(waveQ + r) * QSTR + ks * 32], hi);

  v8f sacc[6];
#pragma unroll
  for (int t = 0; t < 6; ++t) sacc[t] = v8f_zero();
#pragma unroll
  for (int t = 0; t < 6; ++t)
#pragma unroll
    for (int ks = 0; ks < 2; ++ks) {
      v16bf bk = load_frag(&lK[(t * 16 + r) * KSTR + ks * 32], hi);
      sacc[t] = __builtin_amdgcn_wmma_f32_16x16x32_bf16(
          false, aq[ks], false, bk, (short)0, sacc[t], false, false);
    }

  __syncthreads();  // Q reads done before P overwrites same LDS

  const int   segOff[3] = {0, LTXT, LTXT + NIMG};
  const int   segLen[3] = {LTXT, NIMG, NLBL};
  const float segScl[3] = {1.0f, SCALE_IMG, SCALE_LBL};

#pragma unroll
  for (int vi = 0; vi < 8; ++vi) {
    float sc[6], w[6];
#pragma unroll
    for (int t = 0; t < 6; ++t) {
      sc[t] = sacc[t][vi] * 0.125f;  // D^-0.5
      w[t]  = 0.0f;
    }
#pragma unroll
    for (int sg = 0; sg < 3; ++sg) {
      const int off = segOff[sg], len = segLen[sg];
      float m = -3.4e38f;
#pragma unroll
      for (int t = 0; t < 6; ++t) {
        const int kidx = t * 16 + r;
        const bool in = (kidx >= off) && (kidx < off + len);
        m = fmaxf(m, in ? sc[t] : -3.4e38f);
      }
#pragma unroll
      for (int mk = 1; mk < 16; mk <<= 1)
        m = fmaxf(m, __shfl_xor(m, mk, 32));
      float e[6], l = 0.0f;
#pragma unroll
      for (int t = 0; t < 6; ++t) {
        const int kidx = t * 16 + r;
        const bool in = (kidx >= off) && (kidx < off + len);
        e[t] = in ? __expf(sc[t] - m) : 0.0f;
        l += e[t];
      }
#pragma unroll
      for (int mk = 1; mk < 16; mk <<= 1)
        l += __shfl_xor(l, mk, 32);
      const float inv = segScl[sg] / l;
#pragma unroll
      for (int t = 0; t < 6; ++t) {
        const int kidx = t * 16 + r;
        if ((kidx >= off) && (kidx < off + len)) w[t] = e[t] * inv;
      }
    }
    const int qrow = waveQ + vi + hi * 8;
#pragma unroll
    for (int t = 0; t < 6; ++t)
      lQP[qrow * PSTR + t * 16 + r] = f2bf(w[t]);
  }

  // out = P (16x96) * V (96x64)
  v8f oacc[4];
#pragma unroll
  for (int fn = 0; fn < 4; ++fn) oacc[fn] = v8f_zero();
#pragma unroll
  for (int ks = 0; ks < 3; ++ks) {
    v16bf pf = load_frag(&lQP[(waveQ + r) * PSTR + ks * 32], hi);
#pragma unroll
    for (int fn = 0; fn < 4; ++fn) {
      v16bf vf = load_frag(&lVT[(fn * 16 + r) * VSTR + ks * 32], hi);
      oacc[fn] = __builtin_amdgcn_wmma_f32_16x16x32_bf16(
          false, pf, false, vf, (short)0, oacc[fn], false, false);
    }
  }

#pragma unroll
  for (int fn = 0; fn < 4; ++fn) {
    const int d = fn * 16 + r;
#pragma unroll
    for (int vi = 0; vi < 8; ++vi) {
      const long s = (long)b * SEQ + qBase + waveQ + vi + hi * 8;
      O[s * CDIM + cOff + d] = f2bf(oacc[fn][vi]);
    }
  }
}

// ---------------------------------------------------------------------------
// Launch. Workspace (bf16/ush elems):
//  HSb 20.97M | ENCb 0.70M | Wq_t 1.64M | Wo_t 1.64M | 6x W*_t 2.62M |
//  Qb 20.97M | Ab 20.97M | Kb 0.44M | Vb 0.44M   ~= 167 MB total.
// ---------------------------------------------------------------------------
extern "C" void kernel_launch(void* const* d_in, const int* in_sizes, int n_in,
                              void* d_out, int out_size, void* d_ws,
                              size_t ws_size, hipStream_t stream) {
  const float* HS   = (const float*)d_in[0];
  const float* ENC  = (const float*)d_in[1];
  const float* W[8] = {(const float*)d_in[2],  // Wq
                       (const float*)d_in[9],  // Wo
                       (const float*)d_in[3],  // Wk
                       (const float*)d_in[4],  // Wv
                       (const float*)d_in[5],  // Wk_ip
                       (const float*)d_in[6],  // Wv_ip
                       (const float*)d_in[7],  // Wk_lbl
                       (const float*)d_in[8]}; // Wv_lbl
  const float* bo = (const float*)d_in[10];
  float* OUT = (float*)d_out;

  const long nHS = (long)BSZ * SEQ * CDIM;    // 20,971,520
  const long nEN = (long)BSZ * LTOT * DCDIM;  // 696,320
  const long nWs = (long)CDIM * CDIM;         // 1,638,400
  const long nWb = (long)CDIM * DCDIM;        // 2,621,440
  const long nKV = (long)BSZ * LTOT * CDIM;   // 435,200

  unsigned short* p    = (unsigned short*)d_ws;
  unsigned short* HSb  = p;            p += nHS;
  unsigned short* ENCb = p;            p += nEN;
  unsigned short* Wt[8];
  Wt[0] = p; p += nWs;                 // Wq_t  (1280x1280)
  Wt[1] = p; p += nWs;                 // Wo_t
  for (int i = 2; i < 8; ++i) { Wt[i] = p; p += nWb; }  // K/V proj (1280x2048)
  unsigned short* Qb = p;  p += nHS;
  unsigned short* Ab = p;  p += nHS;
  unsigned short* Kb = p;  p += nKV;
  unsigned short* Vb = p;  p += nKV;

  const dim3 blk(256);
  const int M = BSZ * SEQ;  // 16384

  // --- pre-pass: bf16 conversions + weight transposes ---
  cvt_bf16<<<dim3((unsigned)((nHS / 4 + 255) / 256)), blk, 0, stream>>>(
      HS, HSb, nHS / 4);
  cvt_bf16<<<dim3((unsigned)((nEN / 4 + 255) / 256)), blk, 0, stream>>>(
      ENC, ENCb, nEN / 4);
  transpose_cvt<<<dim3(CDIM / 32, CDIM / 32), blk, 0, stream>>>(
      W[0], Wt[0], CDIM, CDIM);
  transpose_cvt<<<dim3(CDIM / 32, CDIM / 32), blk, 0, stream>>>(
      W[1], Wt[1], CDIM, CDIM);
  for (int i = 2; i < 8; ++i)
    transpose_cvt<<<dim3(CDIM / 32, DCDIM / 32), blk, 0, stream>>>(
        W[i], Wt[i], DCDIM, CDIM);

  // --- Q = HS @ Wq  (bf16 out) ---
  gemm_bf16t<<<dim3(M / 128, CDIM / 128), blk, 0, stream>>>(
      HSb, Wt[0], nullptr, Qb, M, CDIM, CDIM, M, M, 0, nullptr, nullptr);

  // --- K/V projections scattered into packed (B,85,C) bf16 buffers ---
  gemm_bf16t<<<dim3(3, CDIM / 128), blk, 0, stream>>>(
      ENCb, Wt[2], nullptr, Kb, BSZ * LTXT, CDIM, DCDIM, LTXT, LTOT, 0,
      nullptr, nullptr);
  gemm_bf16t<<<dim3(3, CDIM / 128), blk, 0, stream>>>(
      ENCb, Wt[3], nullptr, Vb, BSZ * LTXT, CDIM, DCDIM, LTXT, LTOT, 0,
      nullptr, nullptr);
  gemm_bf16t<<<dim3(1, CDIM / 128), blk, 0, stream>>>(
      ENCb, Wt[4], nullptr, Kb, BSZ * NIMG, CDIM, DCDIM, NIMG, LTOT, LTXT,
      nullptr, nullptr);
  gemm_bf16t<<<dim3(1, CDIM / 128), blk, 0, stream>>>(
      ENCb, Wt[5], nullptr, Vb, BSZ * NIMG, CDIM, DCDIM, NIMG, LTOT, LTXT,
      nullptr, nullptr);
  gemm_bf16t<<<dim3(1, CDIM / 128), blk, 0, stream>>>(
      ENCb, Wt[6], nullptr, Kb, BSZ * NLBL, CDIM, DCDIM, NLBL, LTOT,
      LTXT + NIMG, nullptr, nullptr);
  gemm_bf16t<<<dim3(1, CDIM / 128), blk, 0, stream>>>(
      ENCb, Wt[7], nullptr, Vb, BSZ * NLBL, CDIM, DCDIM, NLBL, LTOT,
      LTXT + NIMG, nullptr, nullptr);

  // --- fused triple-segment attention on WMMA ---
  attn_wmma<<<dim3(SEQ / 128, HEADS, BSZ), blk, 0, stream>>>(Qb, Kb, Vb, Ab);

  // --- out = attn @ Wo + bo + HS -> f32 d_out ---
  gemm_bf16t<<<dim3(M / 128, CDIM / 128), blk, 0, stream>>>(
      Ab, Wt[1], OUT, nullptr, M, CDIM, CDIM, M, M, 0, bo, HS);
}
